// MultiHeadAttention_64029372449400
// MI455X (gfx1250) — compile-verified, matrix operand order checked by reading, not run on hardware
//
#include <hip/hip_runtime.h>

typedef __attribute__((ext_vector_type(4)))  __bf16 bf4v;
typedef __attribute__((ext_vector_type(8)))  __bf16 bf8v;
typedef __attribute__((ext_vector_type(16))) __bf16 v16bf;
typedef __attribute__((ext_vector_type(8)))  float  v8f;

#define D_DIM  1024
#define B_ROWS 32768
#define LDK    72      // padded LDS K-stride (elements)

// ---- fp32 -> bf16 tile staging helpers (overload on source type) ----
__device__ __forceinline__ bf4v cvt_load4(const float* p) {
  const float4 f = *(const float4*)p;
  bf4v r;
  r[0] = (__bf16)f.x; r[1] = (__bf16)f.y; r[2] = (__bf16)f.z; r[3] = (__bf16)f.w;
  return r;
}
__device__ __forceinline__ bf4v cvt_load4(const __bf16* p) { return *(const bf4v*)p; }

// Gather one 16x32 bf16 fragment from an LDS tile stored row-major (K contiguous).
// ISA layout: lane holds row M=lane&15; half hi=lane>>4 holds K = hi*8 + {0..7} and 16+hi*8 + {0..7}.
__device__ __forceinline__ v16bf load_frag(const __bf16* lds, int row, int kk, int hi) {
  const __bf16* p = lds + row * LDK + kk * 32 + hi * 8;
  bf8v lo  = *(const bf8v*)p;         // ds_load_b128
  bf8v hi8 = *(const bf8v*)(p + 16);  // ds_load_b128
  return __builtin_shufflevector(lo, hi8, 0,1,2,3,4,5,6,7,8,9,10,11,12,13,14,15);
}

// C = A[M,K] @ W[N,K]^T + bias, 128x128 block tile, BK=64, 8 waves (4 M x 2 N),
// per-wave 2x4 grid of 16x16 f32 accumulators.
// EPI==0: out is bf16.  EPI==1: out is fp32 with fused residual add.
template<typename AT, int EPI>
__launch_bounds__(256)
__global__ void gemm_wmma_kernel(const AT* __restrict__ A, const float* __restrict__ W,
                                 const float* __restrict__ bias,
                                 const float* __restrict__ residual,
                                 void* __restrict__ outp)
{
  __shared__ __bf16 lA[128 * LDK];
  __shared__ __bf16 lB[128 * LDK];

  const int t     = threadIdx.x;
  const int lane  = t & 31;
  const int wave  = t >> 5;
  const int waveM = wave & 3;     // 0..3 -> 32 rows each
  const int waveN = wave >> 2;    // 0..1 -> 64 cols each
  const int lrow  = lane & 15;
  const int hi    = lane >> 4;
  const int bM    = blockIdx.y * 128;
  const int bN    = blockIdx.x * 128;

  v8f acc[2][4] = {};

  for (int kb = 0; kb < D_DIM; kb += 64) {
    // stage 128x64 A tile and 128x64 W tile (converted to bf16)
#pragma unroll
    for (int i = 0; i < 8; ++i) {
      const int c    = t + i * 256;     // 0..2047 4-element chunks
      const int row  = c >> 4;
      const int col4 = (c & 15) * 4;
      *(bf4v*)&lA[row * LDK + col4] = cvt_load4(A + (size_t)(bM + row) * D_DIM + kb + col4);
      *(bf4v*)&lB[row * LDK + col4] = cvt_load4(W + (size_t)(bN + row) * D_DIM + kb + col4);
    }
    __syncthreads();

#pragma unroll
    for (int kk = 0; kk < 2; ++kk) {
      v16bf af[2], bfr[4];
#pragma unroll
      for (int mi = 0; mi < 2; ++mi)
        af[mi] = load_frag(lA, waveM * 32 + mi * 16 + lrow, kk, hi);
#pragma unroll
      for (int ni = 0; ni < 4; ++ni)
        bfr[ni] = load_frag(lB, waveN * 64 + ni * 16 + lrow, kk, hi);
#pragma unroll
      for (int mi = 0; mi < 2; ++mi)
#pragma unroll
        for (int ni = 0; ni < 4; ++ni)
          acc[mi][ni] = __builtin_amdgcn_wmma_f32_16x16x32_bf16(
              false, af[mi], false, bfr[ni], (short)0, acc[mi][ni], false, false);
    }
    __syncthreads();
  }

  // epilogue: C layout -> lane holds col n=lane&15, rows M = hi*8 + v
#pragma unroll
  for (int mi = 0; mi < 2; ++mi) {
#pragma unroll
    for (int ni = 0; ni < 4; ++ni) {
      const int n  = bN + waveN * 64 + ni * 16 + lrow;
      const float bn = bias[n];
#pragma unroll
      for (int v = 0; v < 8; ++v) {
        const int m = bM + waveM * 32 + mi * 16 + hi * 8 + v;
        const size_t idx = (size_t)m * D_DIM + n;
        const float val = acc[mi][ni][v] + bn;
        if constexpr (EPI == 0) {
          ((__bf16*)outp)[idx] = (__bf16)val;
        } else {
          ((float*)outp)[idx] = val + residual[idx];
        }
      }
    }
  }
}

// Per-row head-wise attention: 1 wave per row. energy[h][g] = Q[h].K[g]/8,
// softmax over g, out[h][d] = sum_g attn*V[g][d], stored (d,h)-transposed.
__launch_bounds__(256)
__global__ void attn_kernel(const __bf16* __restrict__ Q, const __bf16* __restrict__ K,
                            const __bf16* __restrict__ V, __bf16* __restrict__ O)
{
  __shared__ __bf16 sQ[8][D_DIM];
  __shared__ __bf16 sK[8][D_DIM];
  __shared__ __bf16 sV[8][D_DIM];
  __shared__ float  sA[8][256];

  const int t = threadIdx.x, lane = t & 31, w = t >> 5;
  const size_t row = (size_t)blockIdx.x * 8 + w;

  const bf8v* gq = (const bf8v*)(Q + row * D_DIM);
  const bf8v* gk = (const bf8v*)(K + row * D_DIM);
  const bf8v* gv = (const bf8v*)(V + row * D_DIM);
  bf8v* q8 = (bf8v*)sQ[w]; bf8v* k8 = (bf8v*)sK[w]; bf8v* v8 = (bf8v*)sV[w];
#pragma unroll
  for (int i = 0; i < 4; ++i) {
    q8[lane + i * 32] = gq[lane + i * 32];
    k8[lane + i * 32] = gk[lane + i * 32];
    v8[lane + i * 32] = gv[lane + i * 32];
  }
  // each wave only touches its own LDS slice -> same-wave DS ordering suffices

  // lane -> fixed head h = lane/2, 8 keys g = (lane&1)*8 + j
  const int h = lane >> 1, g0 = (lane & 1) * 8;
  float e[8];
#pragma unroll
  for (int j = 0; j < 8; ++j) e[j] = 0.0f;
  for (int d = 0; d < 64; ++d) {
    const float qd = (float)sQ[w][h * 64 + d];
#pragma unroll
    for (int j = 0; j < 8; ++j)
      e[j] += qd * (float)sK[w][(g0 + j) * 64 + d];
  }
  float mx = -3.0e38f;
#pragma unroll
  for (int j = 0; j < 8; ++j) { e[j] *= 0.125f; mx = fmaxf(mx, e[j]); }
  mx = fmaxf(mx, __shfl_xor(mx, 1));
  float p[8], s = 0.0f;
#pragma unroll
  for (int j = 0; j < 8; ++j) { p[j] = __expf(e[j] - mx); s += p[j]; }
  s += __shfl_xor(s, 1);
  const float inv = 1.0f / s;
#pragma unroll
  for (int j = 0; j < 8; ++j) sA[w][h * 16 + g0 + j] = p[j] * inv;

  // out: 1024 values per row, 32 per lane; write transposed (d*16 + h)
  for (int tt = 0; tt < 32; ++tt) {
    const int oi = tt * 32 + lane;
    const int hh = oi >> 6, d = oi & 63;
    float acc = 0.0f;
#pragma unroll
    for (int g = 0; g < 16; ++g)
      acc += sA[w][hh * 16 + g] * (float)sV[w][g * 64 + d];
    O[row * D_DIM + d * 16 + hh] = (__bf16)acc;
  }
}

// In-place LayerNorm, one block per row.
__launch_bounds__(256)
__global__ void ln_kernel(float* __restrict__ X, const float* __restrict__ w,
                          const float* __restrict__ b)
{
  __shared__ float r1[256], r2[256];
  const int t = threadIdx.x;
  float* rowp = X + (size_t)blockIdx.x * D_DIM;
  const float4 x = ((const float4*)rowp)[t];
  r1[t] = x.x + x.y + x.z + x.w;
  r2[t] = x.x * x.x + x.y * x.y + x.z * x.z + x.w * x.w;
  __syncthreads();
  for (int o = 128; o > 0; o >>= 1) {
    if (t < o) { r1[t] += r1[t + o]; r2[t] += r2[t + o]; }
    __syncthreads();
  }
  const float mu = r1[0] * (1.0f / D_DIM);
  const float var = r2[0] * (1.0f / D_DIM) - mu * mu;
  const float rs = rsqrtf(var + 1e-5f);
  const float4 w4 = ((const float4*)w)[t];
  const float4 b4 = ((const float4*)b)[t];
  float4 y;
  y.x = (x.x - mu) * rs * w4.x + b4.x;
  y.y = (x.y - mu) * rs * w4.y + b4.y;
  y.z = (x.z - mu) * rs * w4.z + b4.z;
  y.w = (x.w - mu) * rs * w4.w + b4.w;
  ((float4*)rowp)[t] = y;
}

extern "C" void kernel_launch(void* const* d_in, const int* in_sizes, int n_in,
                              void* d_out, int out_size, void* d_ws, size_t ws_size,
                              hipStream_t stream)
{
  (void)in_sizes; (void)n_in; (void)out_size; (void)ws_size;
  const float* sa  = (const float*)d_in[0];
  const float* st  = (const float*)d_in[1];
  const float* Wq  = (const float*)d_in[2];
  const float* bq  = (const float*)d_in[3];
  const float* Wk  = (const float*)d_in[4];
  const float* bk  = (const float*)d_in[5];
  const float* Wv  = (const float*)d_in[6];
  const float* bv  = (const float*)d_in[7];
  const float* Wo  = (const float*)d_in[8];
  const float* bo  = (const float*)d_in[9];
  const float* lnw = (const float*)d_in[10];
  const float* lnb = (const float*)d_in[11];
  float* out = (float*)d_out;

  char* ws = (char*)d_ws;
  const size_t MATB = (size_t)B_ROWS * D_DIM * sizeof(__bf16);  // 64 MiB
  __bf16* Qb = (__bf16*)(ws + 0 * MATB);
  __bf16* Kb = (__bf16*)(ws + 1 * MATB);
  __bf16* Vb = (__bf16*)(ws + 2 * MATB);
  __bf16* Ab = (__bf16*)(ws + 3 * MATB);

  const dim3 gg(D_DIM / 128, B_ROWS / 128);  // (8, 256)
  gemm_wmma_kernel<float, 0><<<gg, 256, 0, stream>>>(sa, Wq, bq, nullptr, Qb);
  gemm_wmma_kernel<float, 0><<<gg, 256, 0, stream>>>(st, Wk, bk, nullptr, Kb);
  gemm_wmma_kernel<float, 0><<<gg, 256, 0, stream>>>(st, Wv, bv, nullptr, Vb);
  attn_kernel<<<B_ROWS / 8, 256, 0, stream>>>(Qb, Kb, Vb, Ab);
  gemm_wmma_kernel<__bf16, 1><<<gg, 256, 0, stream>>>(Ab, Wo, bo, sa, out);
  ln_kernel<<<B_ROWS, 256, 0, stream>>>(out, lnw, lnb);
}
